// Net_83794811945603
// MI455X (gfx1250) — matrix operand order repaired and checked by reading, not compiled, not run on hardware
//
#include <hip/hip_runtime.h>

// ---------------------------------------------------------------------------
// CDNA5 (gfx1250) GraphSAGE forward.
// GEMMs: v_wmma_f32_16x16x32_bf16 (wave32), f32 accumulate.
// Block tile 128x128, BK=32, 8 waves (4 M x 2 N), wave tile 32x64.
// Weights pre-transposed to [Npad x Kpad] bf16 so all LDS staging is b128.
// Interior blocks take a guard-free fast path (only 1/391 M-blocks is edge).
// Edge aggregation: global_atomic_add_f32 via unsafeAtomicAdd.
// ---------------------------------------------------------------------------

typedef __attribute__((ext_vector_type(16))) __bf16 v16bf;
typedef __attribute__((ext_vector_type(8)))  float  v8f;

union FragU { v16bf v; uint4 q[2]; };

// ---------------------------------------------------------------------------
// f32 -> bf16 with zero padding: src[R x C] -> dst[Rp x (1<<CpShift)]
// ---------------------------------------------------------------------------
__global__ __launch_bounds__(256) void convert_pad_kernel(
    const float* __restrict__ src, __bf16* __restrict__ dst,
    int R, int C, int CpShift, long total)
{
    long idx = (long)blockIdx.x * 256 + threadIdx.x;
    if (idx >= total) return;
    int r = (int)(idx >> CpShift);
    int c = (int)(idx & ((1 << CpShift) - 1));
    float v = (r < R && c < C) ? src[(long)r * C + c] : 0.0f;
    dst[idx] = (__bf16)v;
}

// ---------------------------------------------------------------------------
// Weight transpose+pad+convert: src[K x Nn] f32 -> dst[Npad x (1<<KpShift)]
// bf16, dst[n][k] = src[k][n]; zero-filled padding.  (Tiny matrices.)
// ---------------------------------------------------------------------------
__global__ __launch_bounds__(256) void convert_padT_kernel(
    const float* __restrict__ src, __bf16* __restrict__ dst,
    int K, int Nn, int KpShift, long total)
{
    long idx = (long)blockIdx.x * 256 + threadIdx.x;
    if (idx >= total) return;
    int n = (int)(idx >> KpShift);
    int k = (int)(idx & ((1 << KpShift) - 1));
    float v = (k < K && n < Nn) ? src[(long)k * Nn + n] : 0.0f;
    dst[idx] = (__bf16)v;
}

// ---------------------------------------------------------------------------
// Degree count + inverse (mean denominator); edges fixed across layers.
// ---------------------------------------------------------------------------
__global__ __launch_bounds__(256) void degree_kernel(
    const int* __restrict__ dst, float* __restrict__ cnt, int E)
{
    int i = blockIdx.x * 256 + threadIdx.x;
    if (i < E) unsafeAtomicAdd(&cnt[dst[i]], 1.0f);
}

__global__ __launch_bounds__(256) void inv_kernel(float* __restrict__ cnt, int n)
{
    int i = blockIdx.x * 256 + threadIdx.x;
    if (i < n) cnt[i] = 1.0f / fmaxf(cnt[i], 1.0f);
}

// ---------------------------------------------------------------------------
// agg[dst[e], :] += H[src[e], :]   (float4 per thread, f32 atomics in L2)
// ---------------------------------------------------------------------------
__global__ __launch_bounds__(256) void scatter_add_kernel(
    const float* __restrict__ H, const int* __restrict__ src,
    const int* __restrict__ dst, float* __restrict__ agg,
    long total, int dshift, int din)
{
    long idx = (long)blockIdx.x * 256 + threadIdx.x;
    if (idx >= total) return;
    int e = (int)(idx >> dshift);
    int c = (int)(idx & ((1 << dshift) - 1)) << 2;
    int s = src[e], d = dst[e];
    const float4 v = *(const float4*)(H + (size_t)s * din + c);
    float* p = agg + (size_t)d * din + c;
    unsafeAtomicAdd(p + 0, v.x);
    unsafeAtomicAdd(p + 1, v.y);
    unsafeAtomicAdd(p + 2, v.z);
    unsafeAtomicAdd(p + 3, v.w);
}

// ---------------------------------------------------------------------------
// out = bf16(agg * inv_count[row])   (mean, converted for WMMA A operand)
// ---------------------------------------------------------------------------
__global__ __launch_bounds__(256) void mean_bf16_kernel(
    const float* __restrict__ agg, const float* __restrict__ inv,
    __bf16* __restrict__ out, long total, int dshift)
{
    long idx = (long)blockIdx.x * 256 + threadIdx.x;
    if (idx >= total) return;
    int n = (int)(idx >> dshift);
    out[idx] = (__bf16)(agg[idx] * inv[n]);
}

// ---------------------------------------------------------------------------
// bf16 WMMA GEMM: C[M,N](f32) (+)= A[M,K] @ BT[N,K]^T  [+bias][relu][bf16 copy]
// A: bf16, lda >= K, K % 32 == 0 (zero padded).
// BT: bf16, weights pre-transposed+padded to [Npad x Kpad] -> no bounds checks
// in the K loop and pure b128 staging.
// Block tile 128x128, BK=32; 8 waves arranged 4(M) x 2(N); wave tile 32x64.
// ---------------------------------------------------------------------------
template<int BIAS, int ACCUM, int RELU, int BF16OUT>
__global__ __launch_bounds__(256) void gemm_bf16_wmma_kernel(
    const __bf16* __restrict__ A, int lda,
    const __bf16* __restrict__ BT, int ldb,   // ldb = Kpad
    const float*  __restrict__ bias,
    float*        __restrict__ C, int ldc,
    __bf16*       __restrict__ Cbf,
    int M, int N, int K)
{
    constexpr int BM = 128, BN = 128, BK = 32;
    constexpr int AST = BK + 8;   // 40: keeps every 8-elem chunk 16B aligned
    constexpr int BST = BK + 8;
    __shared__ __align__(16) __bf16 As[BM * AST];    // [m][k]
    __shared__ __align__(16) __bf16 Bs[BN * BST];    // [n][k]

    const int tid  = threadIdx.x;
    const int lane = tid & 31;
    const int wave = tid >> 5;
    const int wm   = wave & 3;          // 0..3  (M direction, 32 rows each)
    const int wn   = wave >> 2;         // 0..1  (N direction, 64 cols each)
    const int half = lane >> 4;         // 0/1
    const int l15  = lane & 15;
    const int block_m = blockIdx.x * BM;
    const int block_n = blockIdx.y * BN;
    const bool interior = (block_m + BM <= M);   // block-uniform

    // staging coordinates (2 chunks per thread for each of A, BT)
    const int r0 = tid >> 2;                 // 0..63   (chunk set 0)
    const int r1 = (tid + 256) >> 2;         // 64..127 (chunk set 1)
    const int kc0 = (tid & 3) * 8;           // 0,8,16,24

    v8f acc[2][4];
#pragma unroll
    for (int mi = 0; mi < 2; ++mi)
#pragma unroll
        for (int ni = 0; ni < 4; ++ni) acc[mi][ni] = {};

    for (int k0 = 0; k0 < K; k0 += BK) {
        // ---- stage A (128x32) and BT (128x32): b128 loads + b128 stores ----
        if (interior) {
            const uint4 a0 = *(const uint4*)(A + (size_t)(block_m + r0) * lda + k0 + kc0);
            const uint4 a1 = *(const uint4*)(A + (size_t)(block_m + r1) * lda + k0 + kc0);
            const uint4 b0 = *(const uint4*)(BT + (size_t)(block_n + r0) * ldb + k0 + kc0);
            const uint4 b1 = *(const uint4*)(BT + (size_t)(block_n + r1) * ldb + k0 + kc0);
            *(uint4*)&As[r0 * AST + kc0] = a0;
            *(uint4*)&As[r1 * AST + kc0] = a1;
            *(uint4*)&Bs[r0 * BST + kc0] = b0;
            *(uint4*)&Bs[r1 * BST + kc0] = b1;
        } else {
            uint4 a0 = make_uint4(0u, 0u, 0u, 0u);
            uint4 a1 = make_uint4(0u, 0u, 0u, 0u);
            if (block_m + r0 < M)
                a0 = *(const uint4*)(A + (size_t)(block_m + r0) * lda + k0 + kc0);
            if (block_m + r1 < M)
                a1 = *(const uint4*)(A + (size_t)(block_m + r1) * lda + k0 + kc0);
            const uint4 b0 = *(const uint4*)(BT + (size_t)(block_n + r0) * ldb + k0 + kc0);
            const uint4 b1 = *(const uint4*)(BT + (size_t)(block_n + r1) * ldb + k0 + kc0);
            *(uint4*)&As[r0 * AST + kc0] = a0;
            *(uint4*)&As[r1 * AST + kc0] = a1;
            *(uint4*)&Bs[r0 * BST + kc0] = b0;
            *(uint4*)&Bs[r1 * BST + kc0] = b1;
        }
        __syncthreads();

        // ---- fragments per ISA VGPR layouts ----
        FragU af[2], bfr[4];
#pragma unroll
        for (int mi = 0; mi < 2; ++mi) {
            const int row  = wm * 32 + mi * 16 + l15;
            const int base = row * AST + half * 8;
            af[mi].q[0] = *(const uint4*)&As[base];        // k = 8h   .. 8h+7
            af[mi].q[1] = *(const uint4*)&As[base + 16];   // k = 8h+16.. 8h+23
        }
#pragma unroll
        for (int ni = 0; ni < 4; ++ni) {
            const int n    = wn * 64 + ni * 16 + l15;
            const int base = n * BST + half * 8;
            bfr[ni].q[0] = *(const uint4*)&Bs[base];       // k = 8h   .. 8h+7
            bfr[ni].q[1] = *(const uint4*)&Bs[base + 16];  // k = 8h+16.. 8h+23
        }

#pragma unroll
        for (int mi = 0; mi < 2; ++mi)
#pragma unroll
            for (int ni = 0; ni < 4; ++ni)
                acc[mi][ni] = __builtin_amdgcn_wmma_f32_16x16x32_bf16(
                    false, af[mi].v, false, bfr[ni].v,
                    (short)0, acc[mi][ni], false, false);

        __syncthreads();
    }

    // ---- epilogue: bias / accumulate-into-C / relu / f32 + bf16 stores ----
#pragma unroll
    for (int mi = 0; mi < 2; ++mi) {
#pragma unroll
        for (int ni = 0; ni < 4; ++ni) {
            const int n = block_n + wn * 64 + ni * 16 + l15;
            if (n >= N) continue;
            const int mbase = block_m + wm * 32 + mi * 16 + half * 8;
            if (interior) {
#pragma unroll
                for (int r = 0; r < 8; ++r) {
                    const int m = mbase + r;
                    float v = acc[mi][ni][r];
                    if (BIAS)  v += bias[n];
                    if (ACCUM) v += C[(size_t)m * ldc + n];
                    if (RELU)  v = fmaxf(v, 0.0f);
                    C[(size_t)m * ldc + n] = v;
                    if (BF16OUT) Cbf[(size_t)m * ldc + n] = (__bf16)v;
                }
            } else {
#pragma unroll
                for (int r = 0; r < 8; ++r) {
                    const int m = mbase + r;
                    if (m >= M) continue;
                    float v = acc[mi][ni][r];
                    if (BIAS)  v += bias[n];
                    if (ACCUM) v += C[(size_t)m * ldc + n];
                    if (RELU)  v = fmaxf(v, 0.0f);
                    C[(size_t)m * ldc + n] = v;
                    if (BF16OUT) Cbf[(size_t)m * ldc + n] = (__bf16)v;
                }
            }
        }
    }
}

// ---------------------------------------------------------------------------
// log_softmax over 40 classes, one wave32 per row.
// ---------------------------------------------------------------------------
__global__ __launch_bounds__(256) void log_softmax40_kernel(
    const float* __restrict__ X, float* __restrict__ out, int M)
{
    const int gw   = (blockIdx.x * 256 + threadIdx.x) >> 5;
    const int lane = threadIdx.x & 31;
    if (gw >= M) return;
    const float* row = X + (size_t)gw * 40;
    float v0 = row[lane];
    float v1 = (lane < 8) ? row[32 + lane] : -3.4e38f;
    float m = fmaxf(v0, v1);
#pragma unroll
    for (int o = 16; o > 0; o >>= 1) m = fmaxf(m, __shfl_xor(m, o, 32));
    float s = __expf(v0 - m) + ((lane < 8) ? __expf(v1 - m) : 0.0f);
#pragma unroll
    for (int o = 16; o > 0; o >>= 1) s += __shfl_xor(s, o, 32);
    const float lse = m + __logf(s);
    out[(size_t)gw * 40 + lane] = v0 - lse;
    if (lane < 8) out[(size_t)gw * 40 + 32 + lane] = v1 - lse;
}

// ---------------------------------------------------------------------------
// Host orchestration
// ---------------------------------------------------------------------------
static inline long cdiv(long a, long b) { return (a + b - 1) / b; }

extern "C" void kernel_launch(void* const* d_in, const int* in_sizes, int n_in,
                              void* d_out, int out_size, void* d_ws, size_t ws_size,
                              hipStream_t stream)
{
    constexpr int N    = 50000;
    constexpr int E    = 800000;
    constexpr int F_IN = 500;
    constexpr int KP0  = 512;   // F_IN padded to mult of 32
    constexpr int D1   = 128;   // ID_DIM == HID
    constexpr int D2   = 256;   // 2*HID
    constexpr int NC   = 40;    // N_CLS
    constexpr int NCP  = 128;   // padded

    const float* x     = (const float*)d_in[0];
    const int*   eidx  = (const int*)d_in[1];
    const float* W_map = (const float*)d_in[2];
    const float* b_map = (const float*)d_in[3];
    const float* Wl1   = (const float*)d_in[4];
    const float* bl1   = (const float*)d_in[5];
    const float* Wr1   = (const float*)d_in[6];
    const float* Wl2   = (const float*)d_in[7];
    const float* bl2   = (const float*)d_in[8];
    const float* Wr2   = (const float*)d_in[9];
    const float* Wl3   = (const float*)d_in[10];
    const float* bl3   = (const float*)d_in[11];
    const float* Wr3   = (const float*)d_in[12];
    const int* src = eidx;
    const int* dst = eidx + E;
    float* out = (float*)d_out;

    // ---- workspace carve-out (256B aligned) ----
    char* base = (char*)d_ws;
    size_t off = 0;
    auto carve = [&](size_t bytes) -> char* {
        char* p = base + off;
        off += (bytes + 255) & ~(size_t)255;
        return p;
    };
    __bf16* Xbf    = (__bf16*)carve((size_t)N * KP0 * 2);
    __bf16* WmapbT = (__bf16*)carve((size_t)D1 * KP0 * 2);   // [128 x 512]
    __bf16* Wl1bT  = (__bf16*)carve((size_t)D1 * D1 * 2);    // [128 x 128]
    __bf16* Wr1bT  = (__bf16*)carve((size_t)D1 * D1 * 2);
    __bf16* Wl2bT  = (__bf16*)carve((size_t)D2 * D1 * 2);    // [256 x 128]
    __bf16* Wr2bT  = (__bf16*)carve((size_t)D2 * D1 * 2);
    __bf16* Wl3bT  = (__bf16*)carve((size_t)NCP * D2 * 2);   // [128 x 256]
    __bf16* Wr3bT  = (__bf16*)carve((size_t)NCP * D2 * 2);
    float*  invc   = (float*)carve((size_t)N * 4);
    float*  agg    = (float*)carve((size_t)N * D2 * 4);
    __bf16* meanb  = (__bf16*)carve((size_t)N * D2 * 2);
    float*  Hf_a   = (float*)carve((size_t)N * D2 * 4);
    __bf16* Hbf_a  = (__bf16*)carve((size_t)N * D2 * 2);
    float*  Hf_b   = (float*)carve((size_t)N * D2 * 4);
    __bf16* Hbf_b  = (__bf16*)carve((size_t)N * D2 * 2);
    if (off > ws_size) return;  // workspace too small: bail deterministically

    const int gM = (int)cdiv(N, 128);   // 391 M-blocks

    // ---- 1) convert activations (row-major) / weights (transposed) ----
    {
        long t = (long)N * KP0;
        convert_pad_kernel<<<(int)cdiv(t, 256), 256, 0, stream>>>(x, Xbf, N, F_IN, 9, t);
        t = (long)D1 * KP0;
        convert_padT_kernel<<<(int)cdiv(t, 256), 256, 0, stream>>>(W_map, WmapbT, F_IN, D1, 9, t);
        t = (long)D1 * D1;
        convert_padT_kernel<<<(int)cdiv(t, 256), 256, 0, stream>>>(Wl1, Wl1bT, D1, D1, 7, t);
        convert_padT_kernel<<<(int)cdiv(t, 256), 256, 0, stream>>>(Wr1, Wr1bT, D1, D1, 7, t);
        t = (long)D2 * D1;
        convert_padT_kernel<<<(int)cdiv(t, 256), 256, 0, stream>>>(Wl2, Wl2bT, D1, D2, 7, t);
        convert_padT_kernel<<<(int)cdiv(t, 256), 256, 0, stream>>>(Wr2, Wr2bT, D1, D2, 7, t);
        t = (long)NCP * D2;
        convert_padT_kernel<<<(int)cdiv(t, 256), 256, 0, stream>>>(Wl3, Wl3bT, D2, NC, 8, t);
        convert_padT_kernel<<<(int)cdiv(t, 256), 256, 0, stream>>>(Wr3, Wr3bT, D2, NC, 8, t);
    }

    // ---- 2) degrees -> 1/max(cnt,1) (edges invariant across layers) ----
    hipMemsetAsync(invc, 0, (size_t)N * 4, stream);
    degree_kernel<<<(int)cdiv(E, 256), 256, 0, stream>>>(dst, invc, E);
    inv_kernel<<<(int)cdiv(N, 256), 256, 0, stream>>>(invc, N);

    // ---- 3) H0 = x @ W_map + b_map  (f32 + bf16 mirror) ----
    gemm_bf16_wmma_kernel<1, 0, 0, 1><<<dim3(gM, 1), 256, 0, stream>>>(
        Xbf, KP0, WmapbT, KP0, b_map, Hf_a, D1, Hbf_a, N, D1, KP0);

    // ---- Layer 1: relu(mean @ Wl1 + bl1 + H0 @ Wr1) ----
    hipMemsetAsync(agg, 0, (size_t)N * D1 * 4, stream);
    {
        long t = (long)E * (D1 / 4);
        scatter_add_kernel<<<(int)cdiv(t, 256), 256, 0, stream>>>(Hf_a, src, dst, agg, t, 5, D1);
        long u = (long)N * D1;
        mean_bf16_kernel<<<(int)cdiv(u, 256), 256, 0, stream>>>(agg, invc, meanb, u, 7);
    }
    gemm_bf16_wmma_kernel<1, 0, 0, 0><<<dim3(gM, 1), 256, 0, stream>>>(
        meanb, D1, Wl1bT, D1, bl1, Hf_b, D1, (__bf16*)nullptr, N, D1, D1);
    gemm_bf16_wmma_kernel<0, 1, 1, 1><<<dim3(gM, 1), 256, 0, stream>>>(
        Hbf_a, D1, Wr1bT, D1, (const float*)nullptr, Hf_b, D1, Hbf_b, N, D1, D1);

    // ---- Layer 2: relu(mean @ Wl2 + bl2 + H1 @ Wr2) ----
    hipMemsetAsync(agg, 0, (size_t)N * D1 * 4, stream);
    {
        long t = (long)E * (D1 / 4);
        scatter_add_kernel<<<(int)cdiv(t, 256), 256, 0, stream>>>(Hf_b, src, dst, agg, t, 5, D1);
        long u = (long)N * D1;
        mean_bf16_kernel<<<(int)cdiv(u, 256), 256, 0, stream>>>(agg, invc, meanb, u, 7);
    }
    gemm_bf16_wmma_kernel<1, 0, 0, 0><<<dim3(gM, 2), 256, 0, stream>>>(
        meanb, D1, Wl2bT, D1, bl2, Hf_a, D2, (__bf16*)nullptr, N, D2, D1);
    gemm_bf16_wmma_kernel<0, 1, 1, 1><<<dim3(gM, 2), 256, 0, stream>>>(
        Hbf_b, D1, Wr2bT, D1, (const float*)nullptr, Hf_a, D2, Hbf_a, N, D2, D1);

    // ---- Layer 3: mean @ Wl3 + bl3 + H2 @ Wr3  (no relu) ----
    hipMemsetAsync(agg, 0, (size_t)N * D2 * 4, stream);
    {
        long t = (long)E * (D2 / 4);
        scatter_add_kernel<<<(int)cdiv(t, 256), 256, 0, stream>>>(Hf_a, src, dst, agg, t, 6, D2);
        long u = (long)N * D2;
        mean_bf16_kernel<<<(int)cdiv(u, 256), 256, 0, stream>>>(agg, invc, meanb, u, 8);
    }
    gemm_bf16_wmma_kernel<1, 0, 0, 0><<<dim3(gM, 1), 256, 0, stream>>>(
        meanb, D2, Wl3bT, D2, bl3, Hf_b, NC, (__bf16*)nullptr, N, NC, D2);
    gemm_bf16_wmma_kernel<0, 1, 0, 0><<<dim3(gM, 1), 256, 0, stream>>>(
        Hbf_a, D2, Wr3bT, D2, (const float*)nullptr, Hf_b, NC, (__bf16*)nullptr, N, NC, D2);

    // ---- 4) log_softmax over 40 classes ----
    log_softmax40_kernel<<<(int)cdiv((long)N * 32, 256), 256, 0, stream>>>(Hf_b, out, N);
}
